// MultiHeadAttentionClassical_65481071396642
// MI455X (gfx1250) — matrix operand order repaired and checked by reading, not compiled, not run on hardware
//
#include <hip/hip_runtime.h>
#include <hip/hip_bf16.h>

typedef __attribute__((ext_vector_type(16))) _Float16 v16h;
typedef __attribute__((ext_vector_type(8)))  float    v8f;
typedef int v4i_ __attribute__((vector_size(16)));   // matches builtin's pointee type

#define EMBED 1024
#define HEADS 16
#define DKDIM 64
#define BATCH 4
#define SEQ   1024
#define MROWS (BATCH*SEQ)   // 4096

union H16 { v16h v; uint4 u[2]; };

// ---------------- async global->LDS copy (gfx1250 ASYNCcnt path), with fallback
#if defined(__has_builtin)
#if __has_builtin(__builtin_amdgcn_global_load_async_to_lds_b128) && \
    __has_builtin(__builtin_amdgcn_s_wait_asynccnt)
#define USE_ASYNC_LDS 1
#endif
#endif

__device__ __forceinline__ void cp16(_Float16* l, const _Float16* g) {
#if defined(USE_ASYNC_LDS)
  __builtin_amdgcn_global_load_async_to_lds_b128(
      (__attribute__((address_space(1))) v4i_*)g,
      (__attribute__((address_space(3))) v4i_*)l, 0, 0);
#else
  *(uint4*)l = *(const uint4*)g;
#endif
}

template <int N>
__device__ __forceinline__ void async_wait() {
#if defined(USE_ASYNC_LDS)
  __builtin_amdgcn_s_wait_asynccnt(N);
#endif
}

// ---------------------------------------------------------------- fp32 -> f16
__global__ void cvt_f32_f16(const float* __restrict__ in, _Float16* __restrict__ out, int n) {
  int i = blockIdx.x * blockDim.x + threadIdx.x;
  if (i < n) out[i] = (_Float16)in[i];
}

// ---------------------------------------- W[F][E] fp32 -> Wt[E][F] f16 (K-major B operand)
__global__ void transpose_cvt(const float* __restrict__ W, _Float16* __restrict__ Wt) {
  __shared__ float tile[32][33];
  int f0 = blockIdx.y * 32, e0 = blockIdx.x * 32;
  int tx = threadIdx.x, ty = threadIdx.y;           // block (32,8)
  for (int i = 0; i < 4; i++)
    tile[ty + i * 8][tx] = W[(f0 + ty + i * 8) * EMBED + e0 + tx];
  __syncthreads();
  for (int i = 0; i < 4; i++)
    Wt[(e0 + ty + i * 8) * EMBED + f0 + tx] = (_Float16)tile[tx][ty + i * 8];
}

// ---------------------------------------------------------------- WMMA GEMM
// C[M,N] = A[M,K] (f16 row-major) * B[K,N] (f16 row-major)
// MODE 0: write f16 in [B][H][S][DK] head layout (QKV projections)
// MODE 1: write f32 row-major (final output projection)
#define BM 128
#define BN 128
#define BK 32
#define SA 40    // BK + 8 pad (halves)
#define SB 136   // BN + 8 pad (halves)

template <int MODE>
__global__ __launch_bounds__(256)
void gemm_wmma(const _Float16* __restrict__ A, const _Float16* __restrict__ B,
               void* __restrict__ Cout, int M, int N, int K) {
  __shared__ _Float16 As[2][BM * SA];
  __shared__ _Float16 Bs[2][BK * SB];
  int tid = threadIdx.x;
  int lane = tid & 31, wid = tid >> 5;
  int wm = wid & 3, wn = wid >> 2;                  // 4 x 2 wave grid
  int lo = lane & 15, hi = lane >> 4;
  int m0 = blockIdx.y * BM, n0 = blockIdx.x * BN;

  // per-thread copy coordinates (4 x 16B chunks per tile per thread)
  int ca_row[2], ca_off[2], cb_row[2], cb_off[2];
  for (int i = 0; i < 2; i++) {
    int c = tid + i * 256;
    ca_row[i] = c >> 2;  ca_off[i] = (c & 3) * 8;
    cb_row[i] = c >> 4;  cb_off[i] = (c & 15) * 8;
  }

  v8f acc[2][4];
  for (int mt = 0; mt < 2; mt++)
    for (int nt = 0; nt < 4; nt++)
      for (int r = 0; r < 8; r++) acc[mt][nt][r] = 0.0f;

  // prologue: stage tile 0 into buffer 0
  for (int i = 0; i < 2; i++)
    cp16(&As[0][ca_row[i] * SA + ca_off[i]],
         &A[(long)(m0 + ca_row[i]) * K + ca_off[i]]);
  for (int i = 0; i < 2; i++)
    cp16(&Bs[0][cb_row[i] * SB + cb_off[i]],
         &B[(long)cb_row[i] * N + n0 + cb_off[i]]);

  for (int kb = 0; kb < K; kb += BK) {
    int buf = (kb / BK) & 1;
    if (kb + BK < K) {                    // stage next tile into other buffer
      int kn = kb + BK, nb = buf ^ 1;
      for (int i = 0; i < 2; i++)
        cp16(&As[nb][ca_row[i] * SA + ca_off[i]],
             &A[(long)(m0 + ca_row[i]) * K + kn + ca_off[i]]);
      for (int i = 0; i < 2; i++)
        cp16(&Bs[nb][cb_row[i] * SB + cb_off[i]],
             &B[(long)(kn + cb_row[i]) * N + n0 + cb_off[i]]);
      async_wait<4>();                    // only current tile's 4 ops must land
    } else {
      async_wait<0>();
    }
    __syncthreads();

    H16 afr[2], bfr[4];
    for (int mt = 0; mt < 2; mt++) {
      int r = wm * 32 + mt * 16 + lo;
      int k0 = hi * 8;
      afr[mt].u[0] = *(const uint4*)&As[buf][r * SA + k0];
      afr[mt].u[1] = *(const uint4*)&As[buf][r * SA + k0 + 16];
    }
    for (int nt = 0; nt < 4; nt++) {
      int nb = wn * 64 + nt * 16;
      for (int s = 0; s < 16; s++)
        bfr[nt].v[s] = Bs[buf][(hi * 8 + (s & 7) + (s >> 3) * 16) * SB + nb + lo];
    }
    for (int mt = 0; mt < 2; mt++)
      for (int nt = 0; nt < 4; nt++)
        acc[mt][nt] = __builtin_amdgcn_wmma_f32_16x16x32_f16(
            false, afr[mt].v, false, bfr[nt].v, (short)0, acc[mt][nt], false, false);
    __syncthreads();
  }

  for (int mt = 0; mt < 2; mt++)
    for (int nt = 0; nt < 4; nt++)
      for (int r = 0; r < 8; r++) {
        int mrow = m0 + wm * 32 + mt * 16 + r + hi * 8;
        int ncol = n0 + wn * 64 + nt * 16 + lo;
        float v = acc[mt][nt][r];
        if (MODE == 0) {
          int b = mrow >> 10, s = mrow & 1023;
          int h = ncol >> 6,  d = ncol & 63;
          ((_Float16*)Cout)[(((b * HEADS + h) << 10) + s) * DKDIM + d] = (_Float16)v;
        } else {
          ((float*)Cout)[(long)mrow * N + ncol] = v;
        }
      }
}

// ---------------------------------------------------------------- flash attention
// grid (SEQ/64, HEADS, BATCH), block 128 (4 waves, each owns 16 query rows)
#define QP 72   // 64 + 8 pad (halves)

__global__ __launch_bounds__(128)
void attn_kernel(const _Float16* __restrict__ Q, const _Float16* __restrict__ Kh,
                 const _Float16* __restrict__ Vh, const int* __restrict__ msk,
                 const float* __restrict__ tmask, _Float16* __restrict__ Ctx) {
  __shared__ _Float16 Qs[64 * QP];
  __shared__ _Float16 Kt[64 * QP];     // transposed: Kt[d][sk]
  __shared__ _Float16 Vs[64 * QP];     // natural:    Vs[sk][d]
  __shared__ _Float16 Ps[4 * 16 * QP]; // per-wave P tiles (A-operand layout)

  int tid = threadIdx.x;
  int lane = tid & 31, w = tid >> 5;
  int lo = lane & 15, hi = lane >> 4;
  int q0 = blockIdx.x * 64, h = blockIdx.y, b = blockIdx.z;
  int bh = b * HEADS + h;
  const _Float16* Qp = Q  + (long)bh * SEQ * DKDIM;
  const _Float16* Kp = Kh + (long)bh * SEQ * DKDIM;
  const _Float16* Vp = Vh + (long)bh * SEQ * DKDIM;

  // stage Q block (64 x 64) once (async; completion covered by first wait+barrier)
  for (int i = 0; i < 4; i++) {
    int c = tid + i * 128;
    int row = c >> 3, off = (c & 7) * 8;
    cp16(&Qs[row * QP + off], &Qp[(q0 + row) * DKDIM + off]);
  }

  v8f o[4];
  for (int nt = 0; nt < 4; nt++)
    for (int r = 0; r < 8; r++) o[nt][r] = 0.0f;
  float mprev[8], lsum[8];
  for (int r = 0; r < 8; r++) { mprev[r] = -3.0e38f; lsum[r] = 0.0f; }

  _Float16* Pw = &Ps[w * 16 * QP];

  for (int kb = 0; kb < SEQ; kb += 64) {
    __syncthreads();
    // prefetch next K/V block while we work on this one
    if (kb + 64 < SEQ) {
      __builtin_prefetch(&Kp[(kb + 64) * DKDIM + tid * 32], 0, 0);
      __builtin_prefetch(&Vp[(kb + 64) * DKDIM + tid * 32], 0, 0);
    }
    // stage K (transposed through VGPRs) and V (async) blocks
    for (int i = 0; i < 4; i++) {
      int c = tid + i * 128;
      int row = c >> 3, off = (c & 7) * 8;
      uint4 tmpu = *(const uint4*)&Kp[(kb + row) * DKDIM + off];
      const _Float16* tv = (const _Float16*)&tmpu;
      for (int j = 0; j < 8; j++) Kt[(off + j) * QP + row] = tv[j];
      cp16(&Vs[row * QP + off], &Vp[(kb + row) * DKDIM + off]);
    }
    async_wait<0>();
    __syncthreads();

    // S = Q * K^T   (16 q-rows x 64 keys per wave)
    v8f sc[4];
    for (int nt = 0; nt < 4; nt++)
      for (int r = 0; r < 8; r++) sc[nt][r] = 0.0f;
    for (int ks = 0; ks < 2; ks++) {
      H16 af;
      int qrow = w * 16 + lo;
      int k0 = ks * 32 + hi * 8;
      af.u[0] = *(const uint4*)&Qs[qrow * QP + k0];
      af.u[1] = *(const uint4*)&Qs[qrow * QP + k0 + 16];
      for (int nt = 0; nt < 4; nt++) {
        H16 bf;
        for (int s = 0; s < 16; s++)
          bf.v[s] = Kt[(ks * 32 + hi * 8 + (s & 7) + (s >> 3) * 16) * QP + nt * 16 + lo];
        sc[nt] = __builtin_amdgcn_wmma_f32_16x16x32_f16(
            false, af.v, false, bf.v, (short)0, sc[nt], false, false);
      }
    }

    // scale * train_mask, then masked fill with -1e9 (exact reference order)
    int qgb = q0 + w * 16 + hi * 8;
    for (int nt = 0; nt < 4; nt++) {
      int col = kb + nt * 16 + lo;
      for (int r = 0; r < 8; r++) {
        int qg = qgb + r;
        float sv = sc[nt][r] * 0.125f;          // 1/sqrt(64)
        sv *= tmask[qg * SEQ + col];
        if (msk[(b * SEQ + qg) * SEQ + col] == 0) sv = -1.0e9f;
        sc[nt][r] = sv;
      }
    }

    // online softmax per row (wave halves carry independent rows)
    for (int r = 0; r < 8; r++) {
      float vmax = fmaxf(fmaxf(sc[0][r], sc[1][r]), fmaxf(sc[2][r], sc[3][r]));
      for (int xm = 8; xm >= 1; xm >>= 1)
        vmax = fmaxf(vmax, __shfl_xor(vmax, xm, 32));
      float mnew  = fmaxf(mprev[r], vmax);
      float alpha = __expf(mprev[r] - mnew);
      mprev[r] = mnew;
      float psum = 0.0f;
      for (int nt = 0; nt < 4; nt++) {
        float p = __expf(sc[nt][r] - mnew);
        psum += p;
        Pw[(r + hi * 8) * QP + nt * 16 + lo] = (_Float16)p;
      }
      for (int xm = 8; xm >= 1; xm >>= 1)
        psum += __shfl_xor(psum, xm, 32);
      lsum[r] = lsum[r] * alpha + psum;
      for (int nt = 0; nt < 4; nt++) o[nt][r] *= alpha;
    }

    // O += P * V
    for (int ks = 0; ks < 2; ks++) {
      H16 af;
      int k0 = ks * 32 + hi * 8;
      af.u[0] = *(const uint4*)&Pw[lo * QP + k0];
      af.u[1] = *(const uint4*)&Pw[lo * QP + k0 + 16];
      for (int nt = 0; nt < 4; nt++) {
        H16 bf;
        for (int s = 0; s < 16; s++)
          bf.v[s] = Vs[(ks * 32 + hi * 8 + (s & 7) + (s >> 3) * 16) * QP + nt * 16 + lo];
        o[nt] = __builtin_amdgcn_wmma_f32_16x16x32_f16(
            false, af.v, false, bf.v, (short)0, o[nt], false, false);
      }
    }
  }

  // normalize and write context in [B][S][E] (row-major e) for the final GEMM
  for (int nt = 0; nt < 4; nt++)
    for (int r = 0; r < 8; r++) {
      int qg = q0 + w * 16 + r + hi * 8;
      int d  = nt * 16 + lo;
      float val = o[nt][r] / lsum[r];
      Ctx[((long)b * SEQ + qg) * EMBED + h * DKDIM + d] = (_Float16)val;
    }
}

// ---------------------------------------------------------------- launcher
extern "C" void kernel_launch(void* const* d_in, const int* in_sizes, int n_in,
                              void* d_out, int out_size, void* d_ws, size_t ws_size,
                              hipStream_t stream) {
  const float* x    = (const float*)d_in[0];
  const int*   mask = (const int*)d_in[1];
  const float* Wq   = (const float*)d_in[2];
  const float* Wk   = (const float*)d_in[3];
  const float* Wv   = (const float*)d_in[4];
  const float* Wo   = (const float*)d_in[5];
  const float* tm   = (const float*)d_in[6];

  char* ws = (char*)d_ws;
  _Float16* Xh  = (_Float16*)(ws);
  _Float16* Wqt = (_Float16*)(ws + (size_t)(8)  * (1 << 20));
  _Float16* Wkt = (_Float16*)(ws + (size_t)(10) * (1 << 20));
  _Float16* Wvt = (_Float16*)(ws + (size_t)(12) * (1 << 20));
  _Float16* Wot = (_Float16*)(ws + (size_t)(14) * (1 << 20));
  _Float16* Qh  = (_Float16*)(ws + (size_t)(16) * (1 << 20));
  _Float16* Kh  = (_Float16*)(ws + (size_t)(24) * (1 << 20));
  _Float16* Vh  = (_Float16*)(ws + (size_t)(32) * (1 << 20));
  _Float16* Ctx = (_Float16*)(ws + (size_t)(40) * (1 << 20));

  int nx = MROWS * EMBED;
  cvt_f32_f16<<<dim3((nx + 255) / 256), dim3(256), 0, stream>>>(x, Xh, nx);

  dim3 tb(32, 8), tg(EMBED / 32, EMBED / 32);
  transpose_cvt<<<tg, tb, 0, stream>>>(Wq, Wqt);
  transpose_cvt<<<tg, tb, 0, stream>>>(Wk, Wkt);
  transpose_cvt<<<tg, tb, 0, stream>>>(Wv, Wvt);
  transpose_cvt<<<tg, tb, 0, stream>>>(Wo, Wot);

  dim3 gg(EMBED / BN, MROWS / BM);
  gemm_wmma<0><<<gg, 256, 0, stream>>>(Xh, Wqt, Qh, MROWS, EMBED, EMBED);
  gemm_wmma<0><<<gg, 256, 0, stream>>>(Xh, Wkt, Kh, MROWS, EMBED, EMBED);
  gemm_wmma<0><<<gg, 256, 0, stream>>>(Xh, Wvt, Vh, MROWS, EMBED, EMBED);

  attn_kernel<<<dim3(SEQ / 64, HEADS, BATCH), 128, 0, stream>>>(Qh, Kh, Vh, mask, tm, Ctx);

  gemm_wmma<1><<<gg, 256, 0, stream>>>(Ctx, Wot, d_out, MROWS, EMBED, EMBED);
}
